// HoloGraphBlockV7_58488864637015
// MI455X (gfx1250) — matrix operand order repaired and checked by reading, not compiled, not run on hardware
//
#include <hip/hip_runtime.h>
#include <cstdint>

// ====================== CDNA5 async global->LDS (guarded; fallback = staged copy) ======
#if __has_builtin(__builtin_amdgcn_global_load_async_to_lds_b128) && \
    __has_builtin(__builtin_amdgcn_s_wait_asynccnt)
#define HAVE_ASYNC_LDS 1
#else
#define HAVE_ASYNC_LDS 0
#endif

// Exact parameter type per hipcc diagnostic:
//   __attribute__((vector_size(4*sizeof(int)))) int __device__ *   (AS1, global)
//   and the LDS side in AS3 (__shared__).
typedef int v4i_vs __attribute__((vector_size(4 * sizeof(int))));
typedef __attribute__((address_space(1))) v4i_vs* gvec_ptr;
typedef __attribute__((address_space(3))) v4i_vs* lvec_ptr;

// ====================== bf16 helpers (raw-bit, RNE) ======================
__device__ __forceinline__ uint16_t f2bf(float f) {
    uint32_t u = __builtin_bit_cast(uint32_t, f);
    uint32_t r = u + 0x7FFFu + ((u >> 16) & 1u);
    return (uint16_t)(r >> 16);
}
__device__ __forceinline__ float bf2f(uint16_t h) {
    uint32_t u = ((uint32_t)h) << 16;
    return __builtin_bit_cast(float, u);
}

typedef __attribute__((ext_vector_type(16))) __bf16 v16bf;
typedef __attribute__((ext_vector_type(8)))  float  v8f;

union BF16Frag { v16bf v; uint4 q[2]; };

// Problem constants
constexpr int CB = 4, CT = 2048, CE = 512, CH = 8, CD = 32;
constexpr int CM = CB * CT;          // 8192 token rows
constexpr int CTD = CH * CD;         // 256
constexpr int CFF = 4 * CE;          // 2048

// ====================== fp32 -> bf16 convert ======================
__global__ __launch_bounds__(256) void cvt_bf16_k(const float* __restrict__ in,
                                                  uint16_t* __restrict__ out, int n) {
    int i = blockIdx.x * 256 + threadIdx.x;
    if (i < n) out[i] = f2bf(in[i]);
}

// ====================== LayerNorm -> bf16 (wave32 per row, E=512) ======================
__global__ __launch_bounds__(256) void ln_bf16_k(const float* __restrict__ x,
                                                 const float* __restrict__ g,
                                                 const float* __restrict__ b,
                                                 uint16_t* __restrict__ out, int rows) {
    int wid = threadIdx.x >> 5, lane = threadIdx.x & 31;
    int row = blockIdx.x * 8 + wid;
    if (row >= rows) return;
    const float* xr = x + (size_t)row * CE;
    float vals[16];
    float s = 0.f;
#pragma unroll
    for (int i = 0; i < 16; i++) { vals[i] = xr[lane + i * 32]; s += vals[i]; }
#pragma unroll
    for (int o = 16; o > 0; o >>= 1) s += __shfl_xor(s, o, 32);
    float mean = s * (1.0f / CE);
    float vs = 0.f;
#pragma unroll
    for (int i = 0; i < 16; i++) { float d = vals[i] - mean; vs += d * d; }
#pragma unroll
    for (int o = 16; o > 0; o >>= 1) vs += __shfl_xor(vs, o, 32);
    float inv = rsqrtf(vs * (1.0f / CE) + 1e-5f);
    uint16_t* orow = out + (size_t)row * CE;
#pragma unroll
    for (int i = 0; i < 16; i++) {
        int idx = lane + i * 32;
        float y = (vals[i] - mean) * inv * g[idx] + b[idx];
        orow[idx] = f2bf(y);
    }
}

// ====================== Tiled BF16 WMMA GEMM: C[M,N] = A[M,K] @ W[N,K]^T + bias ======
// Tile: 128(M) x 64(N), K-step 32, DOUBLE-BUFFERED LDS with async global->LDS staging.
// 256 threads = 8 waves in 4x2 grid; each wave: 2x2 v_wmma_f32_16x16x32_bf16 per K-step.
// EPI=0: fp32 out (+bias). EPI=1: exact-GELU then bf16 out (+bias).
template <int EPI>
__global__ __launch_bounds__(256) void gemm_bf16_k(const uint16_t* __restrict__ A,
                                                   const uint16_t* __restrict__ W,
                                                   const float* __restrict__ bias,
                                                   float* __restrict__ Cf,
                                                   uint16_t* __restrict__ Cb,
                                                   int M, int N, int K) {
    __shared__ uint16_t As[2][128 * 32];
    __shared__ uint16_t Wsm[2][64 * 32];
    const int tid = threadIdx.x;
    const int lane = tid & 31;
    const int wave = tid >> 5;
    const int lrow = lane & 15, lhalf = lane >> 4;
    const int wm = wave & 3, wn = wave >> 2;
    const int mBase = blockIdx.y * 128, nBase = blockIdx.x * 64;

    // Per-thread staging coordinates (16B chunks).
    const int ra0 = (tid + 0) >> 2,   ca0 = ((tid + 0) & 3) * 8;
    const int ra1 = (tid + 256) >> 2, ca1 = ((tid + 256) & 3) * 8;
    const int rw  = tid >> 2,         cw  = (tid & 3) * 8;

    auto stage = [&](int buf, int k0) {
#if HAVE_ASYNC_LDS
        // HBM -> LDS through the async engine: no VGPR round-trip, tracked by ASYNCcnt.
        __builtin_amdgcn_global_load_async_to_lds_b128(
            (gvec_ptr)(A + (size_t)(mBase + ra0) * K + k0 + ca0),
            (lvec_ptr)(&As[buf][ra0 * 32 + ca0]), 0, 0);
        __builtin_amdgcn_global_load_async_to_lds_b128(
            (gvec_ptr)(A + (size_t)(mBase + ra1) * K + k0 + ca1),
            (lvec_ptr)(&As[buf][ra1 * 32 + ca1]), 0, 0);
        __builtin_amdgcn_global_load_async_to_lds_b128(
            (gvec_ptr)(W + (size_t)(nBase + rw) * K + k0 + cw),
            (lvec_ptr)(&Wsm[buf][rw * 32 + cw]), 0, 0);
#else
        *(uint4*)(&As[buf][ra0 * 32 + ca0]) =
            *(const uint4*)(A + (size_t)(mBase + ra0) * K + k0 + ca0);
        *(uint4*)(&As[buf][ra1 * 32 + ca1]) =
            *(const uint4*)(A + (size_t)(mBase + ra1) * K + k0 + ca1);
        *(uint4*)(&Wsm[buf][rw * 32 + cw]) =
            *(const uint4*)(W + (size_t)(nBase + rw) * K + k0 + cw);
        if (k0 + 32 < K)
            __builtin_prefetch(A + (size_t)(mBase + (tid >> 1)) * K + (k0 + 32) + (tid & 1) * 16, 0, 1);
#endif
    };

    v8f acc[2][2] = {};

    stage(0, 0);
#if HAVE_ASYNC_LDS
    __builtin_amdgcn_s_wait_asynccnt(0);
#endif
    __syncthreads();

    const int nK = K / 32;
    for (int ki = 0; ki < nK; ki++) {
        const int cur = ki & 1;
        if (ki + 1 < nK) stage(cur ^ 1, (ki + 1) * 32);  // overlap next stage with WMMA

        BF16Frag a[2], b[2];
#pragma unroll
        for (int s = 0; s < 2; s++) {
            // A fragment: lane holds row m, K-chunks {lhalf*8..+7} and {16+lhalf*8..+7}
            int m = wm * 32 + s * 16 + lrow;
            a[s].q[0] = *(const uint4*)(&As[cur][m * 32 + lhalf * 8]);
            a[s].q[1] = *(const uint4*)(&As[cur][m * 32 + 16 + lhalf * 8]);
            // B fragment: lane holds col n, K-chunks {lhalf*8..+7} and {16+lhalf*8..+7}
            int n = wn * 32 + s * 16 + lrow;
            b[s].q[0] = *(const uint4*)(&Wsm[cur][n * 32 + lhalf * 8]);
            b[s].q[1] = *(const uint4*)(&Wsm[cur][n * 32 + 16 + lhalf * 8]);
        }
#pragma unroll
        for (int sm = 0; sm < 2; sm++)
#pragma unroll
            for (int sn = 0; sn < 2; sn++)
                acc[sm][sn] = __builtin_amdgcn_wmma_f32_16x16x32_bf16(
                    false, a[sm].v, false, b[sn].v, (short)0, acc[sm][sn], false, false);

#if HAVE_ASYNC_LDS
        __builtin_amdgcn_s_wait_asynccnt(0);   // next buffer fully landed in LDS
#endif
        __syncthreads();
    }

    // Epilogue: C element (m,n): lane = n%16 + 16*(m%16/8), vgpr r = m%8
#pragma unroll
    for (int sm = 0; sm < 2; sm++) {
#pragma unroll
        for (int sn = 0; sn < 2; sn++) {
            int n = nBase + wn * 32 + sn * 16 + lrow;
            float bv = bias[n];
#pragma unroll
            for (int r = 0; r < 8; r++) {
                int m = mBase + wm * 32 + sm * 16 + lhalf * 8 + r;
                float val = acc[sm][sn][r] + bv;
                size_t off = (size_t)m * N + n;
                if (EPI == 0) {
                    Cf[off] = val;
                } else {
                    float ge = 0.5f * val * (1.0f + erff(val * 0.70710678118654752f));
                    Cb[off] = f2bf(ge);
                }
            }
        }
    }
}

// ====================== Gate projections + activations (d, gw, gf) ======================
// One thread per (token row, head). Writes (B,H,T) layout for the scan.
__global__ __launch_bounds__(256) void gates_k(const uint16_t* __restrict__ xn,
                                               const float* __restrict__ dw, const float* __restrict__ db,
                                               const float* __restrict__ gww, const float* __restrict__ gwb,
                                               const float* __restrict__ gfw, const float* __restrict__ gfb,
                                               float* __restrict__ delta, float* __restrict__ gw2,
                                               float* __restrict__ gf2) {
    int idx = blockIdx.x * 256 + threadIdx.x;     // row*8 + h
    if (idx >= CM * CH) return;
    int row = idx >> 3, h = idx & 7;
    const uint16_t* xr = xn + (size_t)row * CE;
    const float* wd = dw + h * CE;
    const float* ww = gww + h * CE;
    const float* wf = gfw + h * CE;
    float sd = db[h], sw = gwb[h], sf = gfb[h];
    for (int e = 0; e < CE; e++) {
        float xv = bf2f(xr[e]);
        sd += xv * wd[e]; sw += xv * ww[e]; sf += xv * wf[e];
    }
    float dl = (sd > 20.f) ? sd : log1pf(expf(sd));       // softplus
    float sg = 1.f / (1.f + expf(-sw));
    float sq = 1.f / (1.f + expf(-sf));
    int b = row / CT, t = row % CT;
    size_t o = ((size_t)(b * CH + h)) * CT + t;
    delta[o] = dl; gw2[o] = sg * sg; gf2[o] = 1.f - sq * sq;
}

// ====================== Per-head activations: l2norm(k), l2norm(q), tanh(v) ============
// Wave per (row, head); lane = d. Writes (B,H,T,D) layout for the scan.
__global__ __launch_bounds__(256) void head_act_k(const float* __restrict__ kqv,
                                                  float* __restrict__ kT, float* __restrict__ qT,
                                                  float* __restrict__ vT) {
    int wid = threadIdx.x >> 5, lane = threadIdx.x & 31;
    int g = blockIdx.x * 8 + wid;                 // row*8 + h
    if (g >= CM * CH) return;
    int row = g >> 3, h = g & 7;
    const float* K = kqv;
    const float* Q = kqv + (size_t)CM * CTD;
    const float* V = Q + (size_t)CM * CTD;
    size_t src = (size_t)row * CTD + h * CD + lane;
    float kv = K[src], qv = Q[src], vv = V[src];
    float sk = kv * kv, sq = qv * qv;
#pragma unroll
    for (int o = 16; o > 0; o >>= 1) { sk += __shfl_xor(sk, o, 32); sq += __shfl_xor(sq, o, 32); }
    float nk = 1.f / fmaxf(sqrtf(sk), 1e-12f);
    float nq = 1.f / fmaxf(sqrtf(sq), 1e-12f);
    int b = row / CT, t = row % CT;
    size_t dst = (((size_t)(b * CH + h)) * CT + t) * CD + lane;
    kT[dst] = kv * nk; qT[dst] = qv * nq; vT[dst] = tanhf(vv);
}

// ====================== Sequential holographic scan ======================
// One 1024-thread block per (b,h). Thread (d=wave, e=lane) owns state[d][e].
// state_t = decay_t*state + (v*delta*gw)[d]*k[e]; readout[d] = sum_e state*q[e] (wave reduce).
__global__ __launch_bounds__(1024) void scan_k(const float* __restrict__ kT,
                                               const float* __restrict__ qT,
                                               const float* __restrict__ vT,
                                               const float* __restrict__ delta,
                                               const float* __restrict__ gw2,
                                               const float* __restrict__ gf2,
                                               float* __restrict__ ctx,
                                               float* __restrict__ next_mem) {
    int bh = blockIdx.x;                 // 0..31
    int d = threadIdx.x >> 5, e = threadIdx.x & 31;
    int b = bh >> 3, h = bh & 7;
    const float* kp = kT + (size_t)bh * CT * CD;
    const float* qp = qT + (size_t)bh * CT * CD;
    const float* vp = vT + (size_t)bh * CT * CD;
    const float* dp = delta + (size_t)bh * CT;
    const float* gwp = gw2 + (size_t)bh * CT;
    const float* gfp = gf2 + (size_t)bh * CT;
    float s = 0.f;
    for (int t = 0; t < CT; t++) {
        float ke = kp[t * CD + e];
        float qe = qp[t * CD + e];
        float vd = vp[t * CD + d];
        float de = dp[t], gw = gwp[t], gf = gfp[t];
        float decay = fminf(fmaxf(de * gf, 1e-6f), 0.999f);
        float vg = vd * de * gw;
        s = decay * s + vg * ke;
        float r = s * qe;
#pragma unroll
        for (int o = 16; o > 0; o >>= 1) r += __shfl_xor(r, o, 32);
        if (e == 0)
            ctx[(((size_t)b * CT + t) * CH + h) * CD + d] = r;   // (B,T,H*D) row-major
    }
    next_mem[((size_t)bh * CD + d) * CD + e] = s;
}

// ====================== Elementwise glue ======================
__global__ __launch_bounds__(256) void x1_k(const float* __restrict__ x,
                                            const float* __restrict__ oout,
                                            const float* __restrict__ motif,
                                            float* __restrict__ x1, int n) {
    int i = blockIdx.x * 256 + threadIdx.x;
    if (i < n) x1[i] = x[i] + oout[i] + 0.1f * motif[i];
}
__global__ __launch_bounds__(256) void x2_k(const float* __restrict__ x1,
                                            const float* __restrict__ f2out,
                                            float* __restrict__ x2_out,
                                            uint16_t* __restrict__ x2b, int n) {
    int i = blockIdx.x * 256 + threadIdx.x;
    if (i < n) { float v = x1[i] + f2out[i]; x2_out[i] = v; x2b[i] = f2bf(v); }
}
__global__ __launch_bounds__(256) void addps_k(const float* __restrict__ bmin,
                                               const float* __restrict__ bmax,
                                               const float* __restrict__ ps,
                                               float* __restrict__ omin,
                                               float* __restrict__ omax, int n) {
    int i = blockIdx.x * 256 + threadIdx.x;
    if (i < n) { float p = ps[i]; omin[i] = bmin[i] + p; omax[i] = bmax[i] + p; }
}

// ====================== Workspace layout (bytes) ======================
constexpr size_t O_WB   = 0;                                  // bf16 weights (k,q,v,o,ps,f1,f2)
constexpr size_t WB_K   = 0;
constexpr size_t WB_Q   = WB_K + (size_t)CTD * CE;            // elements
constexpr size_t WB_V   = WB_Q + (size_t)CTD * CE;
constexpr size_t WB_O   = WB_V + (size_t)CTD * CE;
constexpr size_t WB_PS  = WB_O + (size_t)CE * CTD;
constexpr size_t WB_F1  = WB_PS + (size_t)CE * CE;
constexpr size_t WB_F2  = WB_F1 + (size_t)CFF * CE;
constexpr size_t WB_END = WB_F2 + (size_t)CE * CFF;           // 2,883,584 elems
constexpr size_t O_XN   = O_WB + WB_END * 2;                  // xn bf16 / x1n bf16 (reused)
constexpr size_t O_KQV  = O_XN + (size_t)CM * CE * 2;         // kqv f32 (24MB) -> o_out -> f2_out
constexpr size_t O_GATE = O_KQV + (size_t)CM * 3 * CTD * 4;   // delta|gw2|gf2 (B,H,T)
constexpr size_t O_KT   = O_GATE + (size_t)3 * CM * CH * 4;   // k,q,v transposed (24MB) -> ps_out
constexpr size_t O_CTX  = O_KT + (size_t)3 * CM * CTD * 4;    // ctx f32 (8MB) -> x2 bf16
constexpr size_t O_CTXB = O_CTX + (size_t)CM * CTD * 4;       // ctx bf16
constexpr size_t O_X1   = O_CTXB + (size_t)CM * CTD * 2;      // x1 f32 (16MB)
constexpr size_t O_HIDB = O_X1 + (size_t)CM * CE * 4;         // hidden bf16 (32MB)

extern "C" void kernel_launch(void* const* d_in, const int* in_sizes, int n_in,
                              void* d_out, int out_size, void* d_ws, size_t ws_size,
                              hipStream_t stream) {
    const float* x     = (const float*)d_in[0];
    const float* b_min = (const float*)d_in[1];
    const float* b_max = (const float*)d_in[2];
    const float* motif = (const float*)d_in[3];
    const float* k_w = (const float*)d_in[4];  const float* k_b = (const float*)d_in[5];
    const float* q_w = (const float*)d_in[6];  const float* q_b = (const float*)d_in[7];
    const float* v_w = (const float*)d_in[8];  const float* v_b = (const float*)d_in[9];
    const float* o_w = (const float*)d_in[10]; const float* o_b = (const float*)d_in[11];
    const float* d_w = (const float*)d_in[12]; const float* d_b = (const float*)d_in[13];
    const float* gw_w= (const float*)d_in[14]; const float* gw_b= (const float*)d_in[15];
    const float* gf_w= (const float*)d_in[16]; const float* gf_b= (const float*)d_in[17];
    const float* ps_w= (const float*)d_in[18]; const float* ps_b= (const float*)d_in[19];
    const float* f1_w= (const float*)d_in[20]; const float* f1_b= (const float*)d_in[21];
    const float* f2_w= (const float*)d_in[22]; const float* f2_b= (const float*)d_in[23];
    const float* ln1_g=(const float*)d_in[24]; const float* ln1_b=(const float*)d_in[25];
    const float* ln2_g=(const float*)d_in[26]; const float* ln2_b=(const float*)d_in[27];

    uint8_t* ws = (uint8_t*)d_ws;
    uint16_t* wb   = (uint16_t*)(ws + O_WB);
    uint16_t* xnB  = (uint16_t*)(ws + O_XN);
    float*    kqv  = (float*)(ws + O_KQV);     // also o_out, then f2_out
    float*    gDel = (float*)(ws + O_GATE);
    float*    gGw  = gDel + (size_t)CM * CH;
    float*    gGf  = gGw  + (size_t)CM * CH;
    float*    kT   = (float*)(ws + O_KT);      // also ps_out later
    float*    qT   = kT + (size_t)CM * CTD;
    float*    vT   = qT + (size_t)CM * CTD;
    float*    ctx  = (float*)(ws + O_CTX);     // also x2 bf16 later
    uint16_t* ctxB = (uint16_t*)(ws + O_CTXB);
    float*    x1   = (float*)(ws + O_X1);
    uint16_t* hidB = (uint16_t*)(ws + O_HIDB);

    float* out_x2   = (float*)d_out;                            // (B,T,E)
    float* out_mem  = out_x2 + (size_t)CM * CE;                 // (B,H,D,D)
    float* out_bmin = out_mem + (size_t)CB * CH * CD * CD;      // (B,T,E)
    float* out_bmax = out_bmin + (size_t)CM * CE;

    const int NTOK = CM * CE;                                   // 4,194,304

    // 1) Weights -> bf16
    auto cvt = [&](const float* src, uint16_t* dst, int n) {
        cvt_bf16_k<<<(n + 255) / 256, 256, 0, stream>>>(src, dst, n);
    };
    cvt(k_w,  wb + WB_K,  CTD * CE);
    cvt(q_w,  wb + WB_Q,  CTD * CE);
    cvt(v_w,  wb + WB_V,  CTD * CE);
    cvt(o_w,  wb + WB_O,  CE * CTD);
    cvt(ps_w, wb + WB_PS, CE * CE);
    cvt(f1_w, wb + WB_F1, CFF * CE);
    cvt(f2_w, wb + WB_F2, CE * CFF);

    // 2) LN1 -> bf16
    ln_bf16_k<<<CM / 8, 256, 0, stream>>>(x, ln1_g, ln1_b, xnB, CM);

    // 3) k,q,v projections (WMMA)
    gemm_bf16_k<0><<<dim3(CTD / 64, CM / 128), 256, 0, stream>>>(
        xnB, wb + WB_K, k_b, kqv, nullptr, CM, CTD, CE);
    gemm_bf16_k<0><<<dim3(CTD / 64, CM / 128), 256, 0, stream>>>(
        xnB, wb + WB_Q, q_b, kqv + (size_t)CM * CTD, nullptr, CM, CTD, CE);
    gemm_bf16_k<0><<<dim3(CTD / 64, CM / 128), 256, 0, stream>>>(
        xnB, wb + WB_V, v_b, kqv + (size_t)2 * CM * CTD, nullptr, CM, CTD, CE);

    // 4) Gate projections + activations
    gates_k<<<(CM * CH + 255) / 256, 256, 0, stream>>>(
        xnB, d_w, d_b, gw_w, gw_b, gf_w, gf_b, gDel, gGw, gGf);

    // 5) Head activations (l2norm k/q, tanh v) into (B,H,T,D)
    head_act_k<<<CM * CH / 8, 256, 0, stream>>>(kqv, kT, qT, vT);

    // 6) Sequential scan -> ctx (B,T,H*D) and next_mem
    scan_k<<<CB * CH, 1024, 0, stream>>>(kT, qT, vT, gDel, gGw, gGf, ctx, out_mem);

    // 7) Output projection
    cvt_bf16_k<<<(CM * CTD + 255) / 256, 256, 0, stream>>>(ctx, ctxB, CM * CTD);
    float* oout = kqv;  // reuse
    gemm_bf16_k<0><<<dim3(CE / 64, CM / 128), 256, 0, stream>>>(
        ctxB, wb + WB_O, o_b, oout, nullptr, CM, CE, CTD);

    // 8) x1 = x + o(ctx) + 0.1*motif
    x1_k<<<(NTOK + 255) / 256, 256, 0, stream>>>(x, oout, motif, x1, NTOK);

    // 9) LN2 -> bf16 (reuse xnB)
    ln_bf16_k<<<CM / 8, 256, 0, stream>>>(x1, ln2_g, ln2_b, xnB, CM);

    // 10) FFN: f1 with fused exact-GELU -> bf16; then f2
    gemm_bf16_k<1><<<dim3(CFF / 64, CM / 128), 256, 0, stream>>>(
        xnB, wb + WB_F1, f1_b, nullptr, hidB, CM, CFF, CE);
    float* f2out = kqv;  // reuse
    gemm_bf16_k<0><<<dim3(CE / 64, CM / 128), 256, 0, stream>>>(
        hidB, wb + WB_F2, f2_b, f2out, nullptr, CM, CE, CFF);

    // 11) x2 = x1 + f2out -> d_out slice 0, plus bf16 copy for ps projection
    uint16_t* x2B = (uint16_t*)ctx;  // reuse ctx region (8MB)
    x2_k<<<(NTOK + 255) / 256, 256, 0, stream>>>(x1, f2out, out_x2, x2B, NTOK);

    // 12) ps projection, then final adds
    float* psout = kT;  // reuse kT region (16MB needed, 24MB available)
    gemm_bf16_k<0><<<dim3(CE / 64, CM / 128), 256, 0, stream>>>(
        x2B, wb + WB_PS, ps_b, psout, nullptr, CM, CE, CE);
    addps_k<<<(NTOK + 255) / 256, 256, 0, stream>>>(
        b_min, b_max, psout, out_bmin, out_bmax, NTOK);

    (void)in_sizes; (void)n_in; (void)out_size; (void)ws_size;
}
